// MoE_23132693856838
// MI455X (gfx1250) — matrix operand order repaired and checked by reading, not compiled, not run on hardware
//
#include <hip/hip_runtime.h>
#include <hip/hip_bf16.h>
#include <math.h>

#define HID 1024
#define ITR 4096
#define NEXP 8

typedef __attribute__((ext_vector_type(16))) __bf16 v16bf;
typedef __attribute__((ext_vector_type(8)))  float  v8f;

union Frag { v16bf v; unsigned u[8]; uint4 q[2]; };

// single-instruction packed f32->bf16x2 (mnemonic confirmed in gfx1250 codegen)
__device__ __forceinline__ unsigned cvt_pk_bf16(float a, float b) {
    unsigned r;
    asm("v_cvt_pk_bf16_f32 %0, %1, %2" : "=v"(r) : "v"(a), "v"(b));
    return r;
}

// Load a 32(K) x 16(N) bf16 B-fragment from a row-major f32 matrix with row
// stride `ld`. Per lane: N = lane%16, h = lane/16; dword v packs
// K = 8h+2v+(v>=4?8:0) and K+1 (table order) -> 16 strided loads + 8x cvt.
__device__ __forceinline__ Frag load_b_frag(const float* __restrict__ tilep, int ld) {
    Frag b;
    const int ln = threadIdx.x & 31;
    const float* p = tilep + (ln & 15);
    const int hh = ln >> 4;
#pragma unroll
    for (int v = 0; v < 8; ++v) {
        const int k = 8 * hh + 2 * v + (v >= 4 ? 8 : 0);
        b.u[v] = cvt_pk_bf16(p[(size_t)k * ld], p[(size_t)(k + 1) * ld]);
    }
    return b;
}

__device__ __forceinline__ float gelu_exact(float v) {
    return 0.5f * v * (1.0f + erff(v * 0.70710678118654752440f));
}

// ---------------------------------------------------------------- gating ----
__global__ void moe_gate(const float* __restrict__ x, const float* __restrict__ Wg,
                         int* __restrict__ top2idx, float* __restrict__ top2w, int T) {
    __shared__ float wg[NEXP * HID];
    const int tid = threadIdx.x;
    for (int i = tid; i < NEXP * HID; i += blockDim.x) wg[i] = Wg[i];
    __syncthreads();
    const int t = blockIdx.x * blockDim.x + tid;
    if (t >= T) return;
    const float* xr = x + (size_t)t * HID;
    float l[NEXP];
#pragma unroll
    for (int e = 0; e < NEXP; ++e) l[e] = 0.f;
    for (int h = 0; h < HID; ++h) {
        const float xv = xr[h];
#pragma unroll
        for (int e = 0; e < NEXP; ++e) l[e] = fmaf(xv, wg[e * HID + h], l[e]);
    }
    int i0 = 0; float v0 = l[0];
#pragma unroll
    for (int e = 1; e < NEXP; ++e) if (l[e] > v0) { v0 = l[e]; i0 = e; }
    int i1 = -1; float v1 = -3.4e38f;
#pragma unroll
    for (int e = 0; e < NEXP; ++e) if (e != i0 && l[e] > v1) { v1 = l[e]; i1 = e; }
    const float e1 = __expf(v1 - v0);   // v0 >= v1
    const float s  = 1.0f + e1;
    top2idx[t * 2 + 0] = i0;  top2w[t * 2 + 0] = 1.0f / s;
    top2idx[t * 2 + 1] = i1;  top2w[t * 2 + 1] = e1 / s;
}

// ------------------------------------------------- deterministic compaction -
__global__ void moe_compact(const int* __restrict__ top2idx, const float* __restrict__ top2w,
                            int* __restrict__ counts, int* __restrict__ list_tok,
                            float* __restrict__ list_w, int T) {
    const int e = blockIdx.x;
    if (threadIdx.x != 0) return;
    int cnt = 0;
    for (int t = 0; t < T; ++t)
        for (int k = 0; k < 2; ++k)
            if (top2idx[t * 2 + k] == e) {
                list_tok[e * T + cnt] = t;
                list_w  [e * T + cnt] = top2w[t * 2 + k];
                ++cnt;
            }
    while (cnt & 15) { list_tok[e * T + cnt] = 0; list_w[e * T + cnt] = 0.f; ++cnt; }
    counts[e] = cnt;
}

// ------------------------------------------------------------------- zero ---
__global__ void moe_zero(float* __restrict__ y, int n) {
    for (int i = blockIdx.x * blockDim.x + threadIdx.x; i < n; i += gridDim.x * blockDim.x)
        y[i] = 0.f;
}

// --------------------------------------------------- fused expert FFN ------
__global__ void __launch_bounds__(256)
moe_ffn(const float* __restrict__ x, const float* __restrict__ W1,
        const float* __restrict__ b1, const float* __restrict__ W2,
        const float* __restrict__ b2, const int* __restrict__ counts,
        const int* __restrict__ list_tok, const float* __restrict__ list_w,
        float* __restrict__ y, int T) {
    const int e = blockIdx.y;
    if (blockIdx.x * 16 >= counts[e]) return;

    __shared__ __bf16 lds_x[16 * 1032];          // row stride 2064B (16B mult)
    __shared__ float  lds_p[8 * 16 * 32];        // per-wave GEMM1 partials
    __shared__ __bf16 lds_h[16 * 40];            // row stride 80B (16B mult)
    __shared__ float  lds_gate[16];
    __shared__ int    lds_tok[16];

    const int tid  = threadIdx.x;
    const int lane = tid & 31;
    const int wv   = tid >> 5;          // 0..7
    const int mrow = lane & 15;
    const int lh   = lane >> 4;

    if (tid < 16) {
        lds_tok [tid] = list_tok[e * T + blockIdx.x * 16 + tid];
        lds_gate[tid] = list_w  [e * T + blockIdx.x * 16 + tid];
    }
    __syncthreads();

    // x tile -> bf16 LDS; read float2, pack once, store b32
    for (int i = tid; i < 16 * (HID / 2); i += 256) {
        const int r = i >> 9, c = (i & 511) * 2;
        const float2 f = *(const float2*)&x[(size_t)lds_tok[r] * HID + c];
        *(unsigned*)&lds_x[r * 1032 + c] = cvt_pk_bf16(f.x, f.y);
    }
    __syncthreads();

    // Precompute 4 A-fragments of x for this wave's K range: 2x ds_load_b128 each.
    Frag aF[4];
#pragma unroll
    for (int ks = 0; ks < 4; ++ks) {
        const int kb = wv * 128 + ks * 32;
        aF[ks].q[0] = *(const uint4*)&lds_x[mrow * 1032 + kb + lh * 8];        // K=kb+lh*8..+7
        aF[ks].q[1] = *(const uint4*)&lds_x[mrow * 1032 + kb + 16 + lh * 8];   // +16
    }

    const float* W1e = W1 + (size_t)e * HID * ITR;
    const float* W2e = W2 + (size_t)e * ITR * HID;

    v8f acc[8];
#pragma unroll
    for (int t = 0; t < 8; ++t) acc[t] = (v8f){};

    for (int i0 = 0; i0 < ITR; i0 += 32) {
        if (i0 + 32 < ITR)
            __builtin_prefetch(W1e + (size_t)(wv * 128 + lane) * ITR + i0 + 32, 0, 1);

        // ---- GEMM1 partials: wave's K range x h-chunk [16,32] ----
#pragma unroll
        for (int ns = 0; ns < 2; ++ns) {
            v8f c = (v8f){};
#pragma unroll
            for (int ks = 0; ks < 4; ++ks) {
                const int gk = wv * 128 + ks * 32;              // B tile K base
                Frag b = load_b_frag(W1e + (size_t)gk * ITR + i0 + ns * 16, ITR);
                c = __builtin_amdgcn_wmma_f32_16x16x32_bf16(false, aF[ks].v, false, b.v,
                                                            (short)0, c, false, false);
            }
#pragma unroll
            for (int r = 0; r < 8; ++r)
                lds_p[wv * 512 + (r + 8 * lh) * 32 + ns * 16 + mrow] = c[r];
        }
        __syncthreads();

        // ---- reduce over waves, +b1, exact GELU -> bf16 h chunk ----
        for (int i = tid; i < 512; i += 256) {
            const int m = i >> 5, n = i & 31;
            float s = b1[e * ITR + i0 + n];
#pragma unroll
            for (int w = 0; w < 8; ++w) s += lds_p[w * 512 + i];
            lds_h[m * 40 + n] = (__bf16)gelu_exact(s);
        }
        __syncthreads();

        // ---- GEMM2: acc[16 x 128 per wave] += h_chunk @ W2[chunk,:] ----
        Frag a2;
        a2.q[0] = *(const uint4*)&lds_h[mrow * 40 + lh * 8];
        a2.q[1] = *(const uint4*)&lds_h[mrow * 40 + 16 + lh * 8];
#pragma unroll
        for (int t = 0; t < 8; ++t) {
            Frag b = load_b_frag(W2e + (size_t)i0 * HID + wv * 128 + t * 16, HID);
            acc[t] = __builtin_amdgcn_wmma_f32_16x16x32_bf16(false, a2.v, false, b.v,
                                                             (short)0, acc[t], false, false);
        }
        __syncthreads();
    }

    // ---- epilogue: y[tok] += gate * (acc + b2); <=2 commutative fp32 adds
    //      per element onto a zeroed buffer -> deterministic ----
#pragma unroll
    for (int t = 0; t < 8; ++t) {
        const int N = wv * 128 + t * 16 + mrow;
        const float bb = b2[e * HID + N];
#pragma unroll
        for (int r = 0; r < 8; ++r) {
            const int M = r + 8 * lh;
            const float g = lds_gate[M];
            if (g != 0.0f)
                unsafeAtomicAdd(&y[(size_t)lds_tok[M] * HID + N], g * (acc[t][r] + bb));
        }
    }
}

// ------------------------------------------------------------------ launch --
extern "C" void kernel_launch(void* const* d_in, const int* in_sizes, int n_in,
                              void* d_out, int out_size, void* d_ws, size_t ws_size,
                              hipStream_t stream) {
    const float* x  = (const float*)d_in[0];
    const float* Wg = (const float*)d_in[1];
    const float* W1 = (const float*)d_in[2];
    const float* b1 = (const float*)d_in[3];
    const float* W2 = (const float*)d_in[4];
    const float* b2 = (const float*)d_in[5];
    float* y = (float*)d_out;

    const int T = in_sizes[0] / HID;           // 2048

    char* ws = (char*)d_ws;
    int*   top2idx = (int*)  (ws);                       // T*2 ints
    float* top2w   = (float*)(ws + 8ull  * T);           // T*2 floats
    int*   counts  = (int*)  (ws + 16ull * T);           // 8 ints (64B slot)
    int*   list_tok= (int*)  (ws + 16ull * T + 64);      // 8*T ints
    float* list_w  = (float*)(ws + 16ull * T + 64 + 32ull * T);

    moe_zero<<<256, 256, 0, stream>>>(y, out_size);
    moe_gate<<<(T + 255) / 256, 256, 0, stream>>>(x, Wg, top2idx, top2w, T);
    moe_compact<<<NEXP, 32, 0, stream>>>(top2idx, top2w, counts, list_tok, list_w, T);
    dim3 grid(T / 16, NEXP);
    moe_ffn<<<grid, 256, 0, stream>>>(x, W1, b1, W2, b2, counts, list_tok, list_w, y, T);
}
